// MambaEncoder_23605140259330
// MI455X (gfx1250) — compile-verified, matrix operand order checked by reading, not run on hardware
//
#include <hip/hip_runtime.h>
#include <hip/hip_bf16.h>

typedef __attribute__((ext_vector_type(16))) _Float16 v16h;
typedef __attribute__((ext_vector_type(8)))  float    v8f;

#define TOKENS   16384      // B*L = 4*4096
#define SEQLEN   4096
#define NBATCH   4
#define DMDIM    256
#define DIDIM    512
#define DSTATE   16
#define DTRANK   16
#define XDBLW    48         // DT_RANK + 2*D_STATE

// ---------------------------------------------------------------------------
// f32 -> f16 conversion (grid-stride)
// ---------------------------------------------------------------------------
__global__ void cvt_f16_kernel(const float* __restrict__ src,
                               _Float16* __restrict__ dst, size_t n) {
    size_t i = (size_t)blockIdx.x * blockDim.x + threadIdx.x;
    size_t stride = (size_t)gridDim.x * blockDim.x;
    for (; i < n; i += stride) dst[i] = (_Float16)src[i];
}

// ---------------------------------------------------------------------------
// WMMA GEMM: C[M,N](f32) = A[M,K](f16,row) * W[N,K](f16,row)^T
// NSUB = compile-time count of 16-wide N tiles per wave (no divergent WMMA).
// wave tile: 32(M) x 16*NSUB(N); two A fragments share every B fragment.
// K loop manually unrolled x2 with independent register sets (no rotation
// copies): all loads for both half-steps issue up front, first WMMA chain
// overlaps the second chain's loads via partial s_wait_loadcnt.
// block = 256 threads = 8 waves arranged 4(M) x 2(N) -> block tile 128 x 128.
// flags bit0: += bias[n]   flags bit1: C += result (residual accumulate)
// Requires K % 64 == 0 (true for K = 128 / 256 / 512).
// ---------------------------------------------------------------------------
template <int NSUB>
__global__ __launch_bounds__(256)
void gemm_wmma_f16(const _Float16* __restrict__ A, const _Float16* __restrict__ W,
                   const float* __restrict__ bias, float* __restrict__ C,
                   int M, int N, int K, int flags)
{
    const int lane = threadIdx.x & 31;
    const int wave = threadIdx.x >> 5;          // 0..7
    const int wm   = wave & 3;                  // M sub-tile in block
    const int wn   = wave >> 2;                 // N group (64 wide each)
    const int m0   = blockIdx.x * 128 + wm * 32;
    const int n0   = blockIdx.y * 128 + wn * 64;
    if (n0 >= N) return;                        // wave-uniform scalar branch
    if (m0 >= M) return;                        // wave-uniform scalar branch

    const int row0  = m0 + (lane & 15);
    const int khalf = (lane >> 4) * 16;
    const int colb  = n0 + (lane & 15);

    const _Float16* aBase = A + (size_t)row0 * K + khalf;
    const _Float16* wBase = W + (size_t)colb * K + khalf;
    const size_t aStride16 = (size_t)16 * K;

    v8f acc0[NSUB];
    v8f acc1[NSUB];
#pragma unroll
    for (int j = 0; j < NSUB; ++j) { acc0[j] = (v8f)(0.f); acc1[j] = (v8f)(0.f); }

    for (int k0 = 0; k0 < K; k0 += 64) {
        // ---- issue every load for both half-steps up front ----
        v16h a0A = *(const v16h*)(aBase + k0);
        v16h a1A = *(const v16h*)(aBase + aStride16 + k0);
        v16h bA[NSUB];
#pragma unroll
        for (int j = 0; j < NSUB; ++j)
            bA[j] = *(const v16h*)(wBase + (size_t)(j * 16) * K + k0);

        v16h a0B = *(const v16h*)(aBase + k0 + 32);
        v16h a1B = *(const v16h*)(aBase + aStride16 + k0 + 32);
        v16h bB[NSUB];
#pragma unroll
        for (int j = 0; j < NSUB; ++j)
            bB[j] = *(const v16h*)(wBase + (size_t)(j * 16) * K + k0 + 32);

        __builtin_prefetch(aBase + k0 + 64, 0, 1);      // global_prefetch_b8

        // ---- WMMA chain A (overlaps B-step loads still in flight) ----
#pragma unroll
        for (int j = 0; j < NSUB; ++j) {
            acc0[j] = __builtin_amdgcn_wmma_f32_16x16x32_f16(
                false, a0A, false, bA[j], (short)0, acc0[j], false, false);
            acc1[j] = __builtin_amdgcn_wmma_f32_16x16x32_f16(
                false, a1A, false, bA[j], (short)0, acc1[j], false, false);
        }
        // ---- WMMA chain B ----
#pragma unroll
        for (int j = 0; j < NSUB; ++j) {
            acc0[j] = __builtin_amdgcn_wmma_f32_16x16x32_f16(
                false, a0B, false, bB[j], (short)0, acc0[j], false, false);
            acc1[j] = __builtin_amdgcn_wmma_f32_16x16x32_f16(
                false, a1B, false, bB[j], (short)0, acc1[j], false, false);
        }
    }

    const int hi = (lane >> 4) * 8;             // M offset of this half-wave
    const int lo = lane & 15;                   // N lane within tile
#pragma unroll
    for (int j = 0; j < NSUB; ++j) {
        int n = n0 + j * 16 + lo;
        float bv = (flags & 1) ? bias[n] : 0.f;
#pragma unroll
        for (int r = 0; r < 8; ++r) {
            int m = m0 + hi + r;
            size_t idx0 = (size_t)m * N + n;
            size_t idx1 = (size_t)(m + 16) * N + n;
            float v0 = acc0[j][r] + bv;
            float v1 = acc1[j][r] + bv;
            if (flags & 2) { v0 += C[idx0]; v1 += C[idx1]; }
            C[idx0] = v0;
            C[idx1] = v1;
        }
    }
}

// ---------------------------------------------------------------------------
// causal depthwise conv (width 4) + bias + SiLU over xs = xz[:, :512]
// writes f32 xu and f16 copy for the next GEMM
// ---------------------------------------------------------------------------
__global__ __launch_bounds__(256)
void conv_silu_kernel(const float* __restrict__ xz, const float* __restrict__ cw,
                      const float* __restrict__ cb, float* __restrict__ xu,
                      _Float16* __restrict__ xuh)
{
    int idx = blockIdx.x * blockDim.x + threadIdx.x;
    if (idx >= TOKENS * DIDIM) return;
    int c = idx & (DIDIM - 1);
    int t = idx >> 9;
    int l = t & (SEQLEN - 1);
    float s = 0.f;
#pragma unroll
    for (int j = 0; j < 4; ++j) {
        int dl = l - 3 + j;
        if (dl >= 0)
            s += cw[c * 4 + j] * xz[(size_t)(t - 3 + j) * 1024 + c];
    }
    s += cb[c];
    float v = s / (1.f + __expf(-s));           // silu
    xu[idx]  = v;
    xuh[idx] = (_Float16)v;
}

// ---------------------------------------------------------------------------
// dt = softplus(x_dbl[:, :16] @ dt_w^T + dt_b) ; one block per token
// ---------------------------------------------------------------------------
__global__ __launch_bounds__(256)
void dt_kernel(const float* __restrict__ xdbl, const float* __restrict__ dtw,
               const float* __restrict__ dtb, float* __restrict__ dt)
{
    __shared__ float r[DTRANK];
    int t = blockIdx.x;
    if (threadIdx.x < DTRANK) r[threadIdx.x] = xdbl[(size_t)t * XDBLW + threadIdx.x];
    __syncthreads();
    for (int c = threadIdx.x; c < DIDIM; c += 256) {
        float s = dtb[c];
#pragma unroll
        for (int j = 0; j < DTRANK; ++j) s += r[j] * dtw[c * DTRANK + j];
        float sp = (s > 20.f) ? s : __logf(1.f + __expf(s));
        dt[(size_t)t * DIDIM + c] = sp;
    }
}

// ---------------------------------------------------------------------------
// selective scan: lanes 0-15 = 16 states of one channel, lanes 16-31 = next.
// 8 waves/block -> 16 channels/block; grid (DI/16, B).
// fuses y += xu*D and y *= silu(z); writes f16 y for out_proj GEMM.
// ---------------------------------------------------------------------------
__global__ __launch_bounds__(256)
void scan_kernel(const float* __restrict__ dt, const float* __restrict__ xdbl,
                 const float* __restrict__ xu, const float* __restrict__ xz,
                 const float* __restrict__ A_log, const float* __restrict__ Dp,
                 _Float16* __restrict__ yh)
{
    int lane = threadIdx.x & 31;
    int wave = threadIdx.x >> 5;
    int s    = lane & 15;
    int ch   = blockIdx.x * 16 + ((wave << 1) | (lane >> 4));
    int b    = blockIdx.y;

    float A = -__expf(A_log[ch * DSTATE + s]);
    float D = Dp[ch];
    float h = 0.f;
    size_t tbase = (size_t)b * SEQLEN;

    for (int l = 0; l < SEQLEN; ++l) {
        size_t t = tbase + l;
        float dtc = dt[t * DIDIM + ch];
        float xc  = xu[t * DIDIM + ch];
        float Bv  = xdbl[t * XDBLW + DTRANK + s];
        float Cv  = xdbl[t * XDBLW + DTRANK + DSTATE + s];
        float dA  = __expf(dtc * A);
        h = h * dA + (dtc * xc) * Bv;
        float y = h * Cv;
        y += __shfl_xor(y, 1, 32);
        y += __shfl_xor(y, 2, 32);
        y += __shfl_xor(y, 4, 32);
        y += __shfl_xor(y, 8, 32);
        if (s == 0) {
            float yy = y + xc * D;
            float z  = xz[t * 1024 + DIDIM + ch];
            float sil = z / (1.f + __expf(-z));
            yh[t * DIDIM + ch] = (_Float16)(yy * sil);
        }
    }
}

// ---------------------------------------------------------------------------
// layernorm over D (256 or 512); one block (256 threads) per token
// ---------------------------------------------------------------------------
template <int D>
__global__ __launch_bounds__(256)
void layernorm_kernel(const float* __restrict__ in, const float* __restrict__ g,
                      const float* __restrict__ bt, float* __restrict__ out)
{
    __shared__ float red[8];
    constexpr int PER = D / 256;
    size_t t = blockIdx.x;
    float vals[PER];
    float sum = 0.f;
#pragma unroll
    for (int i = 0; i < PER; ++i) {
        vals[i] = in[t * D + threadIdx.x + i * 256];
        sum += vals[i];
    }
    for (int off = 16; off > 0; off >>= 1) sum += __shfl_xor(sum, off, 32);
    int wave = threadIdx.x >> 5;
    if ((threadIdx.x & 31) == 0) red[wave] = sum;
    __syncthreads();
    float tot = 0.f;
#pragma unroll
    for (int i = 0; i < 8; ++i) tot += red[i];
    float mean = tot / (float)D;
    __syncthreads();
    float vs = 0.f;
#pragma unroll
    for (int i = 0; i < PER; ++i) { float d = vals[i] - mean; vs += d * d; }
    for (int off = 16; off > 0; off >>= 1) vs += __shfl_xor(vs, off, 32);
    if ((threadIdx.x & 31) == 0) red[wave] = vs;
    __syncthreads();
    float vtot = 0.f;
#pragma unroll
    for (int i = 0; i < 8; ++i) vtot += red[i];
    float rstd = rsqrtf(vtot / (float)D + 1e-5f);
#pragma unroll
    for (int i = 0; i < PER; ++i) {
        int j = threadIdx.x + i * 256;
        out[t * D + j] = (vals[i] - mean) * rstd * g[j] + bt[j];
    }
}

// reverse sequence copy (per batch), DM elems per token
__global__ void reverse_copy_kernel(const float* __restrict__ src, float* __restrict__ dst)
{
    int idx = blockIdx.x * blockDim.x + threadIdx.x;
    if (idx >= TOKENS * DMDIM) return;
    int j = idx & (DMDIM - 1);
    int t = idx >> 8;
    int b = t >> 12, l = t & (SEQLEN - 1);
    dst[idx] = src[((size_t)(b * SEQLEN + (SEQLEN - 1 - l))) * DMDIM + j];
}

// concat [h_fwd | reverse(h_bwd)] -> (tokens, 512)
__global__ void concat_rev_kernel(const float* __restrict__ hf,
                                  const float* __restrict__ hb, float* __restrict__ out)
{
    int idx = blockIdx.x * blockDim.x + threadIdx.x;
    if (idx >= TOKENS * 512) return;
    int j = idx & 511;
    int t = idx >> 9;
    int b = t >> 12, l = t & (SEQLEN - 1);
    float v;
    if (j < DMDIM) v = hf[(size_t)t * DMDIM + j];
    else v = hb[((size_t)(b * SEQLEN + (SEQLEN - 1 - l))) * DMDIM + (j - DMDIM)];
    out[idx] = v;
}

// ---------------------------------------------------------------------------
extern "C" void kernel_launch(void* const* d_in, const int* in_sizes, int n_in,
                              void* d_out, int out_size, void* d_ws, size_t ws_size,
                              hipStream_t stream)
{
    const float* x        = (const float*)d_in[0];
    const float* embed_w  = (const float*)d_in[1];
    const float* embed_b  = (const float*)d_in[2];
    const float* norm_g   = (const float*)d_in[3];
    const float* norm_b   = (const float*)d_in[4];
    const float* in_w     = (const float*)d_in[5];
    const float* conv_w   = (const float*)d_in[6];
    const float* conv_b   = (const float*)d_in[7];
    const float* xp_w     = (const float*)d_in[8];
    const float* dt_w     = (const float*)d_in[9];
    const float* dt_b     = (const float*)d_in[10];
    const float* A_log    = (const float*)d_in[11];
    const float* Dparm    = (const float*)d_in[12];
    const float* out_w    = (const float*)d_in[13];
    const float* onorm_g  = (const float*)d_in[14];
    const float* onorm_b  = (const float*)d_in[15];
    float* out = (float*)d_out;

    // carve workspace
    char* p = (char*)d_ws;
    auto alloc = [&](size_t bytes) {
        void* r = (void*)p;
        p += (bytes + 255) & ~(size_t)255;
        return r;
    };
    _Float16* x_h   = (_Float16*)alloc((size_t)TOKENS * 128 * 2);
    _Float16* ew_h  = (_Float16*)alloc((size_t)DMDIM * 128 * 2);
    _Float16* inw_h = (_Float16*)alloc((size_t)8 * 1024 * DMDIM * 2);
    _Float16* xpw_h = (_Float16*)alloc((size_t)8 * XDBLW * DIDIM * 2);
    _Float16* ow_h  = (_Float16*)alloc((size_t)8 * DMDIM * DIDIM * 2);
    float*    h_fwd = (float*)alloc((size_t)TOKENS * DMDIM * 4);
    float*    h_bwd = (float*)alloc((size_t)TOKENS * DMDIM * 4);
    _Float16* hx    = (_Float16*)alloc((size_t)TOKENS * DMDIM * 2);
    float*    xz    = (float*)alloc((size_t)TOKENS * 1024 * 4);   // also reused as concat buf
    float*    xu    = (float*)alloc((size_t)TOKENS * DIDIM * 4);
    _Float16* xu_h  = (_Float16*)alloc((size_t)TOKENS * DIDIM * 2);
    float*    xdbl  = (float*)alloc((size_t)TOKENS * XDBLW * 4);
    float*    dt    = (float*)alloc((size_t)TOKENS * DIDIM * 4);
    _Float16* y_h   = (_Float16*)alloc((size_t)TOKENS * DIDIM * 2);
    (void)ws_size; (void)n_in; (void)in_sizes; (void)out_size;

    // --- one-time weight/input downconversion to f16 ---
    auto cvt = [&](const float* s, _Float16* d, size_t n) {
        int blocks = (int)((n + 255) / 256);
        if (blocks > 4096) blocks = 4096;
        hipLaunchKernelGGL(cvt_f16_kernel, dim3(blocks), dim3(256), 0, stream, s, d, n);
    };
    cvt(x,       x_h,   (size_t)TOKENS * 128);
    cvt(embed_w, ew_h,  (size_t)DMDIM * 128);
    cvt(in_w,    inw_h, (size_t)8 * 1024 * DMDIM);
    cvt(xp_w,    xpw_h, (size_t)8 * XDBLW * DIDIM);
    cvt(out_w,   ow_h,  (size_t)8 * DMDIM * DIDIM);

    // --- embed: xz(:, :256) = x @ embed_w^T + b; LN -> h_fwd ---
    hipLaunchKernelGGL((gemm_wmma_f16<4>), dim3(TOKENS / 128, 2), dim3(256), 0, stream,
                       x_h, ew_h, embed_b, xz, TOKENS, DMDIM, 128, 1);
    hipLaunchKernelGGL((layernorm_kernel<DMDIM>), dim3(TOKENS), dim3(256), 0, stream,
                       xz, norm_g, norm_b, h_fwd);
    hipLaunchKernelGGL(reverse_copy_kernel, dim3((TOKENS * DMDIM + 255) / 256), dim3(256),
                       0, stream, h_fwd, h_bwd);

    // --- 8 mamba layers: 0..3 on h_fwd, 4..7 on h_bwd ---
    for (int li = 0; li < 8; ++li) {
        float* hcur = (li < 4) ? h_fwd : h_bwd;
        const _Float16* inw = inw_h + (size_t)li * 1024 * DMDIM;
        const _Float16* xpw = xpw_h + (size_t)li * XDBLW * DIDIM;
        const _Float16* ow  = ow_h  + (size_t)li * DMDIM * DIDIM;

        cvt(hcur, hx, (size_t)TOKENS * DMDIM);

        // xz = hx @ in_w^T   (16384 x 1024, K=256)
        hipLaunchKernelGGL((gemm_wmma_f16<4>), dim3(TOKENS / 128, 8), dim3(256), 0, stream,
                           hx, inw, nullptr, xz, TOKENS, 1024, DMDIM, 0);

        // depthwise conv + silu
        hipLaunchKernelGGL(conv_silu_kernel, dim3((TOKENS * DIDIM + 255) / 256), dim3(256),
                           0, stream, xz, conv_w + (size_t)li * DIDIM * 4,
                           conv_b + (size_t)li * DIDIM, xu, xu_h);

        // x_dbl = xu @ xp_w^T   (16384 x 48, K=512) — NSUB=3 covers all 48 cols
        hipLaunchKernelGGL((gemm_wmma_f16<3>), dim3(TOKENS / 128, 1), dim3(256), 0, stream,
                           xu_h, xpw, nullptr, xdbl, TOKENS, XDBLW, DIDIM, 0);

        // dt = softplus(x_dbl[:, :16] @ dt_w^T + dt_b)
        hipLaunchKernelGGL(dt_kernel, dim3(TOKENS), dim3(256), 0, stream,
                           xdbl, dt_w + (size_t)li * DIDIM * DTRANK,
                           dt_b + (size_t)li * DIDIM, dt);

        // selective scan (fused +xu*D and *silu(z))
        hipLaunchKernelGGL(scan_kernel, dim3(DIDIM / 16, NBATCH), dim3(256), 0, stream,
                           dt, xdbl, xu, xz,
                           A_log + (size_t)li * DIDIM * DSTATE,
                           Dparm + (size_t)li * DIDIM, y_h);

        // hcur += y @ out_w^T   (16384 x 256, K=512) — residual fused via accumulate
        hipLaunchKernelGGL((gemm_wmma_f16<4>), dim3(TOKENS / 128, 2), dim3(256), 0, stream,
                           y_h, ow, nullptr, hcur, TOKENS, DMDIM, DIDIM, 2);
    }

    // --- concat [fwd | reversed bwd] then final layernorm(512) ---
    hipLaunchKernelGGL(concat_rev_kernel, dim3((TOKENS * 512 + 255) / 256), dim3(256),
                       0, stream, h_fwd, h_bwd, xz);
    hipLaunchKernelGGL((layernorm_kernel<512>), dim3(TOKENS), dim3(256), 0, stream,
                       xz, onorm_g, onorm_b, out);
}
